// SimplePointTransformerSeg_18923625906664
// MI455X (gfx1250) — compile-verified, matrix operand order checked by reading, not run on hardware
//
#include <hip/hip_runtime.h>
#include <stddef.h>

// ---------------------------------------------------------------------------
// SimplePointTransformerSeg forward for MI455X (gfx1250, wave32, WMMA).
//
// Every conv1d/conv2d in the network is a GEMM  D[o,l] = sum_c W[o,c] X[c,l]
// routed through one WMMA kernel (v_wmma_f32_16x16x32_bf16, bf16 operands
// staged+converted in LDS as packed K-pairs, f32 accumulate).  KNN / FPS /
// BN / gathers / softmax are specialized VALU kernels with 3D grids so no
// runtime integer division is needed anywhere.
//
// Input flattening assumption (jax tree_leaves => sorted dict keys,
// recursively).  Top inputs dict {'params','pc'} -> params first, pc last.
//   params keys sorted: b1, b2, b3, down, in, out, up
//   blk sorted:   W1,W2,bn1_b,bn1_g,bn2_b,bn2_g,bn_b,bn_g,tr(18)
//   layer sorted: Wk,Wq,Wv,aW1,aW2,ab2,abn1_b,abn1_g,abn2_b,abn2_g,
//                 bk,bq,bv,pW1,pW2,pb2,pbn_b,pbn_g
//   in/down/up sorted: W1,W2,bn1_b,bn1_g,bn2_b,bn2_g
//   out sorted:   W1,W2,b2,bn_b,bn_g
// => b1=0, b2=26, b3=52, down=78, in=84, out=90, up=95, pc=101
// ---------------------------------------------------------------------------

#define B_    2
#define N_    4096
#define N2_   1024
#define KNN_  16
#define H_    64
#define NCLS_ 13

typedef __attribute__((ext_vector_type(16))) __bf16 v16bf;
typedef __attribute__((ext_vector_type(8)))  float  v8f;

union BF16x16 {
  v16bf v;
  unsigned int d[8];
  unsigned short u[16];
};

__device__ __forceinline__ unsigned int f2bf(float f) {
  unsigned int x = __float_as_uint(f);
  x += 0x7FFFu + ((x >> 16) & 1u);   // round-to-nearest-even
  return x >> 16;
}

// ---------------------------------------------------------------------------
// Universal WMMA GEMM:  per batch b:  D[o, l] = sum_c W[o,c] * X[b,c,l] (+bias)
//   W: (M,K) fp32 row-major.  X batch stride CX*L (X may be a channel-subset
//   of a CX-channel tensor).  D batch stride MD*L.
// Block = 256 threads = 8 wave32s; block tile = 64(M) x 128(L).
// Wave (wm,wn) owns rows [16*wm,16*wm+16) x cols [64*wn,64*wn+64)
// => 4 accumulators of 16x16 f32.  KT = #32-wide k-tiles (compile-time).
// FULLM = (M==64): branch-free epilogue.
//
// LDS tiles hold packed bf16 K-pairs (one dword = elements k,k+1) so every
// fragment register is one aligned ds_load_b32:
//   Ws32[o * KP2 + k/2]      (A: 16x32, lane = row,  dword idx = packed K)
//   Xs32[(k/2) * 128 + col]  (B: 32x16, lane = col)
// Fragment dword map per ISA 16-bit layout: kd = kt*16 + 4*t + j + (j>=4?4:0)
// ---------------------------------------------------------------------------
template <int KT, bool FULLM>
__global__ __launch_bounds__(256)
void k_gemm_wmma(const float* __restrict__ W, const float* __restrict__ X,
                 const float* __restrict__ bias, float* __restrict__ D,
                 int K, int L, int CX, int M, int MD)
{
  constexpr int KP2 = KT * 16;              // packed (dword) K extent
  __shared__ unsigned int Ws32[64 * KP2];   // [o][k/2]
  __shared__ unsigned int Xs32[KP2 * 128];  // [k/2][col]
  __shared__ float bs[64];
  const int tid = threadIdx.x;
  const int b   = blockIdx.z;
  const int l0  = blockIdx.x * 128;

  const float* Xb = X + (size_t)b * CX * L + l0;

  // stage bias (zeros when absent) -- keeps the epilogue branch-free
  if (tid < 64) bs[tid] = (bias != nullptr && tid < M) ? bias[tid] : 0.0f;
  // stage W (zero-pad rows >= M / cols >= K folded into the expression)
  for (int i = tid; i < 64 * KP2; i += 256) {
    const int o  = i / KP2;
    const int kp = i - o * KP2;
    const int c0 = 2 * kp;
    unsigned int lo = (o < M && c0     < K) ? f2bf(W[o * K + c0])     : 0u;
    unsigned int hi = (o < M && c0 + 1 < K) ? f2bf(W[o * K + c0 + 1]) : 0u;
    Ws32[i] = lo | (hi << 16);
  }
  // stage X tile (pairs of channel rows packed per dword)
  for (int i = tid; i < KP2 * 128; i += 256) {
    const int kp  = i >> 7;
    const int col = i & 127;
    const int c0  = 2 * kp;
    unsigned int lo = (c0     < K) ? f2bf(Xb[(size_t)c0 * L + col])       : 0u;
    unsigned int hi = (c0 + 1 < K) ? f2bf(Xb[(size_t)(c0 + 1) * L + col]) : 0u;
    Xs32[i] = lo | (hi << 16);
    // warm GL2 for the neighboring tile of this streaming activation row
    if (c0 < K) __builtin_prefetch(Xb + (size_t)c0 * L + col + 128, 0, 1);
  }
  __syncthreads();

  const int wave = tid >> 5;
  const int lane = tid & 31;
  const int wm = wave & 3;      // m-tile 0..3
  const int wn = wave >> 2;     // n-half 0..1
  const int t  = lane >> 4;     // lane group
  const int mn = lane & 15;     // row (A) / col (B) within 16x16 tile

  v8f acc[4];
#pragma unroll
  for (int i = 0; i < 4; ++i)
#pragma unroll
    for (int e = 0; e < 8; ++e) acc[i][e] = 0.0f;

  const int arow = wm * 16 + mn;
#pragma unroll
  for (int kt = 0; kt < KT; ++kt) {
    BF16x16 af;
#pragma unroll
    for (int j = 0; j < 8; ++j) {
      const int kd = kt * 16 + 4 * t + j + ((j >= 4) ? 4 : 0);
      af.d[j] = Ws32[arow * KP2 + kd];
    }
#pragma unroll
    for (int nt = 0; nt < 4; ++nt) {
      const int ncol = wn * 64 + nt * 16 + mn;
      BF16x16 bfm;
#pragma unroll
      for (int j = 0; j < 8; ++j) {
        const int kd = kt * 16 + 4 * t + j + ((j >= 4) ? 4 : 0);
        bfm.d[j] = Xs32[kd * 128 + ncol];
      }
      acc[nt] = __builtin_amdgcn_wmma_f32_16x16x32_bf16(
          false, af.v, false, bfm.v, (short)0, acc[nt], false, false);
    }
  }

  // epilogue: bias from LDS, row pointers advanced by += L (no per-row mul)
  const int obase = wm * 16 + t * 8;       // C/D layout: row = r + 8*t
  float bv[8];
#pragma unroll
  for (int r = 0; r < 8; ++r) bv[r] = bs[obase + r];

  float* rowbase = D + (size_t)b * MD * L + (size_t)obase * L + l0;
#pragma unroll
  for (int nt = 0; nt < 4; ++nt) {
    float* ptr = rowbase + wn * 64 + nt * 16 + mn;
    if (FULLM) {
#pragma unroll
      for (int r = 0; r < 8; ++r) { *ptr = acc[nt][r] + bv[r]; ptr += L; }
    } else {
#pragma unroll
      for (int r = 0; r < 8; ++r) {
        if (obase + r < M) *ptr = acc[nt][r] + bv[r];
        ptr += L;
      }
    }
  }
}

// ---------------------------------------------------------------------------
// Training-mode batch norm: per-channel mean/var over (B, L) -> fused
// scale/shift.  One block per channel; nested loops (no division).
// ---------------------------------------------------------------------------
__global__ __launch_bounds__(256)
void k_bn_stats(const float* __restrict__ X, const float* __restrict__ g,
                const float* __restrict__ bt, float* __restrict__ stats,
                int C, int L, int Bn)
{
  const int c = blockIdx.x;
  __shared__ float s1[256], s2[256];
  float sum = 0.f, sq = 0.f;
  for (int bb = 0; bb < Bn; ++bb) {
    const float* Xb = X + ((size_t)bb * C + c) * L;
    for (int l = threadIdx.x; l < L; l += 256) {
      float v = Xb[l];
      sum += v; sq += v * v;
    }
  }
  s1[threadIdx.x] = sum; s2[threadIdx.x] = sq;
  __syncthreads();
  for (int s = 128; s > 0; s >>= 1) {
    if (threadIdx.x < s) {
      s1[threadIdx.x] += s1[threadIdx.x + s];
      s2[threadIdx.x] += s2[threadIdx.x + s];
    }
    __syncthreads();
  }
  if (threadIdx.x == 0) {
    float inv  = 1.0f / ((float)Bn * (float)L);
    float mean = s1[0] * inv;
    float var  = s2[0] * inv - mean * mean;
    float sc   = g[c] * rsqrtf(var + 1e-5f);
    stats[2 * c]     = sc;
    stats[2 * c + 1] = bt[c] - mean * sc;
  }
}

// grid (ceil(L/256), C, B)
__global__ __launch_bounds__(256)
void k_bn_apply(const float* __restrict__ X, const float* __restrict__ stats,
                const float* __restrict__ res, float* __restrict__ Y,
                int C, int L, int relu)
{
  const int l = blockIdx.x * 256 + threadIdx.x;
  if (l >= L) return;
  const int c = blockIdx.y;
  const size_t i = ((size_t)blockIdx.z * C + c) * L + l;
  float v = stats[2 * c] * X[i] + stats[2 * c + 1];
  if (res)  v += res[i];
  if (relu) v = fmaxf(v, 0.f);
  Y[i] = v;
}

// ---------------------------------------------------------------------------
// KNN: one thread per query point, refs tiled through LDS, register
// insertion-sort top-KQ (smallest squared distances).
// ---------------------------------------------------------------------------
template <int KQ>
__global__ __launch_bounds__(256)
void k_knn(const float* __restrict__ q, const float* __restrict__ ref,
           int* __restrict__ oidx, float* __restrict__ odist, int M, int Nr)
{
  const int b = blockIdx.z;
  const int m = blockIdx.x * 256 + threadIdx.x;
  const float* qb = q   + (size_t)b * M  * 3;
  const float* rb = ref + (size_t)b * Nr * 3;
  float qx = 0.f, qy = 0.f, qz = 0.f;
  if (m < M) { qx = qb[m * 3]; qy = qb[m * 3 + 1]; qz = qb[m * 3 + 2]; }
  float dk[KQ]; int ik[KQ];
#pragma unroll
  for (int s = 0; s < KQ; ++s) { dk[s] = 3.4e38f; ik[s] = 0; }

  __shared__ float rs[256 * 3];
  for (int j0 = 0; j0 < Nr; j0 += 256) {
    __syncthreads();
    const int j = j0 + threadIdx.x;
    if (j < Nr) {
      rs[threadIdx.x * 3]     = rb[j * 3];
      rs[threadIdx.x * 3 + 1] = rb[j * 3 + 1];
      rs[threadIdx.x * 3 + 2] = rb[j * 3 + 2];
    }
    __syncthreads();
    const int cnt = (Nr - j0 < 256) ? (Nr - j0) : 256;
    for (int jj = 0; jj < cnt; ++jj) {
      float dx = qx - rs[jj * 3];
      float dy = qy - rs[jj * 3 + 1];
      float dz = qz - rs[jj * 3 + 2];
      float d  = dx * dx + dy * dy + dz * dz;
      if (d < dk[KQ - 1]) {
        dk[KQ - 1] = d; ik[KQ - 1] = j0 + jj;
#pragma unroll
        for (int s = KQ - 1; s > 0; --s) {
          if (dk[s] < dk[s - 1]) {
            float td = dk[s]; dk[s] = dk[s - 1]; dk[s - 1] = td;
            int   ti = ik[s]; ik[s] = ik[s - 1]; ik[s - 1] = ti;
          }
        }
      }
    }
  }
  if (m < M) {
#pragma unroll
    for (int s = 0; s < KQ; ++s) {
      oidx [((size_t)b * M + m) * KQ + s] = ik[s];
      odist[((size_t)b * M + m) * KQ + s] = dk[s];
    }
  }
}

// ---------------------------------------------------------------------------
// Farthest-point sampling: one block per batch, serial Mout-step scan.
// ---------------------------------------------------------------------------
__global__ __launch_bounds__(256)
void k_fps(const float* __restrict__ p, float* __restrict__ p2,
           int Np, int Mout)
{
  const int b = blockIdx.x;
  __shared__ float dd[N_];
  __shared__ float rv[256];
  __shared__ int   ri[256];
  __shared__ int   s_last;
  const float* pb  = p  + (size_t)b * Np * 3;
  float*       out = p2 + (size_t)b * Mout * 3;

  for (int i = threadIdx.x; i < Np; i += 256) dd[i] = 1e10f;
  if (threadIdx.x == 0) s_last = 0;
  __syncthreads();

  for (int t = 0; t < Mout; ++t) {
    const int last = s_last;
    const float lx = pb[last * 3], ly = pb[last * 3 + 1], lz = pb[last * 3 + 2];
    if (threadIdx.x == 0) { out[t * 3] = lx; out[t * 3 + 1] = ly; out[t * 3 + 2] = lz; }
    float bm = -1.f; int bi = 0;
    for (int i = threadIdx.x; i < Np; i += 256) {
      float dx = pb[i * 3] - lx, dy = pb[i * 3 + 1] - ly, dz = pb[i * 3 + 2] - lz;
      float nd = fminf(dd[i], dx * dx + dy * dy + dz * dz);
      dd[i] = nd;
      if (nd > bm) { bm = nd; bi = i; }
    }
    rv[threadIdx.x] = bm; ri[threadIdx.x] = bi;
    __syncthreads();
    for (int s = 128; s > 0; s >>= 1) {
      if (threadIdx.x < s && rv[threadIdx.x + s] > rv[threadIdx.x]) {
        rv[threadIdx.x] = rv[threadIdx.x + s];
        ri[threadIdx.x] = ri[threadIdx.x + s];
      }
      __syncthreads();
    }
    if (threadIdx.x == 0) s_last = ri[0];
    __syncthreads();
  }
}

// ---------------------------------------------------------------------------
// group(): xyz channels (relative coords) + gathered feature channels.
// ---------------------------------------------------------------------------
// grid (ceil(Mq*16/256), 1, B)
__global__ __launch_bounds__(256)
void k_group_xyz(const float* __restrict__ p, const float* __restrict__ np,
                 const int* __restrict__ idx, float* __restrict__ out,
                 int Mq, int Np, int C3)
{
  const int i = blockIdx.x * 256 + threadIdx.x;   // m*16 + k
  if (i >= Mq * 16) return;
  const int b = blockIdx.z;
  const int m = i >> 4;
  const int j = idx[(size_t)b * Mq * 16 + i];
  const size_t chan = (size_t)Mq * 16;
  const size_t ob = (size_t)b * C3 * chan + i;
  const float* pj  = p  + ((size_t)b * Np + j) * 3;
  const float* npm = np + ((size_t)b * Mq + m) * 3;
  out[ob]            = pj[0] - npm[0];
  out[ob + chan]     = pj[1] - npm[1];
  out[ob + 2 * chan] = pj[2] - npm[2];
}

// grid (ceil(Mq*16/256), C, B)
__global__ __launch_bounds__(256)
void k_group_feat(const float* __restrict__ f, const int* __restrict__ idx,
                  float* __restrict__ out, int C, int C3, int coff,
                  int Mq, int Np)
{
  const int mk = blockIdx.x * 256 + threadIdx.x;
  if (mk >= Mq * 16) return;
  const int c = blockIdx.y;
  const int b = blockIdx.z;
  const int j = idx[(size_t)b * Mq * 16 + mk];
  out[((size_t)b * C3 + coff + c) * ((size_t)Mq * 16) + mk] =
      f[((size_t)b * C + c) * Np + j];
}

// a[b,c,n,k] = q[b,c,n] - nk[b,3+c,n,k] + nr[b,c,n,k]
// grid (ceil(L/256), C, B), L = Npt*16
__global__ __launch_bounds__(256)
void k_attn_form(const float* __restrict__ q, const float* __restrict__ nk,
                 const float* __restrict__ nr, float* __restrict__ A,
                 int C, int C3, int Npt)
{
  const int L = Npt * 16;
  const int l = blockIdx.x * 256 + threadIdx.x;
  if (l >= L) return;
  const int c = blockIdx.y;
  const int b = blockIdx.z;
  const size_t i = ((size_t)b * C + c) * L + l;
  A[i] = q[((size_t)b * C + c) * Npt + (l >> 4)]
       - nk[((size_t)b * C3 + 3 + c) * L + l]
       + nr[i];
}

__global__ __launch_bounds__(256)
void k_add_inplace(float* __restrict__ Y, const float* __restrict__ X,
                   long long total)
{
  long long i = (long long)blockIdx.x * 256 + threadIdx.x;
  if (i < total) Y[i] += X[i];
}

// softmax over k=16 then weighted sum with values (float4 vectorized)
__global__ __launch_bounds__(256)
void k_softmax_dot(const float* __restrict__ A, const float* __restrict__ V,
                   float* __restrict__ O, long long total /*B*C*Npt*/)
{
  long long i = (long long)blockIdx.x * 256 + threadIdx.x;
  if (i >= total) return;
  const float4* a4 = (const float4*)(A + i * 16);
  const float4* v4 = (const float4*)(V + i * 16);
  float a[16], v[16];
#pragma unroll
  for (int p = 0; p < 4; ++p) {
    float4 ta = a4[p], tv = v4[p];
    a[4 * p] = ta.x; a[4 * p + 1] = ta.y; a[4 * p + 2] = ta.z; a[4 * p + 3] = ta.w;
    v[4 * p] = tv.x; v[4 * p + 1] = tv.y; v[4 * p + 2] = tv.z; v[4 * p + 3] = tv.w;
  }
  float mx = a[0];
#pragma unroll
  for (int k = 1; k < 16; ++k) mx = fmaxf(mx, a[k]);
  float s = 0.f, o = 0.f;
#pragma unroll
  for (int k = 0; k < 16; ++k) {
    float e = __expf(a[k] - mx);
    s += e; o += e * v[k];
  }
  O[i] = o / s;
}

__global__ __launch_bounds__(256)
void k_max_k(const float* __restrict__ X, float* __restrict__ Y,
             long long total /*B*C*N2*/)
{
  long long i = (long long)blockIdx.x * 256 + threadIdx.x;
  if (i >= total) return;
  const float4* x4 = (const float4*)(X + i * 16);
  float m = -3.4e38f;
#pragma unroll
  for (int p = 0; p < 4; ++p) {
    float4 t = x4[p];
    m = fmaxf(m, fmaxf(fmaxf(t.x, t.y), fmaxf(t.z, t.w)));
  }
  Y[i] = m;
}

// 3-NN inverse-distance interpolation + skip: O = f2 + sum_j w_j f1[:,idx_j]
// grid (ceil(Nf/256), C, B)
__global__ __launch_bounds__(256)
void k_up_interp(const float* __restrict__ f1, const float* __restrict__ f2,
                 const int* __restrict__ idx3, const float* __restrict__ d3,
                 float* __restrict__ O, int C, int Nf, int Nc)
{
  const int n = blockIdx.x * 256 + threadIdx.x;
  if (n >= Nf) return;
  const int c = blockIdx.y;
  const int b = blockIdx.z;
  const size_t qb = ((size_t)b * Nf + n) * 3;
  float w0 = 1.0f / (fmaxf(d3[qb],     0.f) + 1e-8f);
  float w1 = 1.0f / (fmaxf(d3[qb + 1], 0.f) + 1e-8f);
  float w2 = 1.0f / (fmaxf(d3[qb + 2], 0.f) + 1e-8f);
  const float inv = 1.0f / (w0 + w1 + w2);
  const float* f1b = f1 + ((size_t)b * C + c) * Nc;
  float ux = (w0 * f1b[idx3[qb]] + w1 * f1b[idx3[qb + 1]] +
              w2 * f1b[idx3[qb + 2]]) * inv;
  const size_t i = ((size_t)b * C + c) * Nf + n;
  O[i] = f2[i] + ux;
}

// pc (B,N,19) -> p1 (B,N,3)
__global__ __launch_bounds__(256)
void k_split_p(const float* __restrict__ pc, float* __restrict__ p1,
               long long total /*B*N*/)
{
  long long i = (long long)blockIdx.x * 256 + threadIdx.x;
  if (i >= total) return;
  p1[i * 3]     = pc[i * 19];
  p1[i * 3 + 1] = pc[i * 19 + 1];
  p1[i * 3 + 2] = pc[i * 19 + 2];
}

// pc (B,N,19) -> x0 (B,16,N)   grid (ceil(N/256), 16, B)
__global__ __launch_bounds__(256)
void k_split_x(const float* __restrict__ pc, float* __restrict__ x0, int Npt)
{
  const int n = blockIdx.x * 256 + threadIdx.x;
  if (n >= Npt) return;
  const int c = blockIdx.y;
  const int b = blockIdx.z;
  x0[((size_t)b * 16 + c) * Npt + n] = pc[((size_t)b * Npt + n) * 19 + 3 + c];
}

// ---------------------------------------------------------------------------
// Host orchestration
// ---------------------------------------------------------------------------
static inline int nb256(long long total) { return (int)((total + 255) / 256); }

extern "C" void kernel_launch(void* const* d_in, const int* in_sizes, int n_in,
                              void* d_out, int out_size, void* d_ws, size_t ws_size,
                              hipStream_t stream)
{
  (void)in_sizes; (void)n_in; (void)out_size; (void)ws_size;

  // -- parameter index map (sorted / jax-pytree order) ----------------------
  const int PB1 = 0, PB2 = 26, PB3 = 52, PDW = 78, PIN = 84, POUT = 90, PUP = 95;
  const int PPC = 101;
  // blk offsets:   +0 W1 +1 W2 +2 bn1_b +3 bn1_g +4 bn2_b +5 bn2_g +6 bn_b +7 bn_g +8 tr
  // layer offsets: +0 Wk +1 Wq +2 Wv +3 aW1 +4 aW2 +5 ab2 +6 abn1_b +7 abn1_g
  //                +8 abn2_b +9 abn2_g +10 bk +11 bq +12 bv +13 pW1 +14 pW2
  //                +15 pb2 +16 pbn_b +17 pbn_g
  auto P = [&](int i) { return (const float*)d_in[i]; };

  // -- workspace layout (floats) -------------------------------------------
  float* WS = (float*)d_ws;
  size_t off = 0;
  auto take = [&](size_t n) { size_t r = off; off += n; return r; };
  const size_t o_p1    = take((size_t)B_ * N_ * 3);
  const size_t o_p2    = take((size_t)B_ * N2_ * 3);
  const size_t o_x0    = take((size_t)B_ * 16 * N_);
  const size_t o_act   = take((size_t)16 * B_ * H_ * N_);     // 16 activation slots
  const size_t o_stats = take(256);
  const size_t o_nk    = take((size_t)B_ * 67 * N_ * 16);
  const size_t o_nv    = take((size_t)B_ * 64 * N_ * 16);
  const size_t o_nr    = take((size_t)B_ * 64 * N_ * 16);
  const size_t o_aa    = take((size_t)B_ * 64 * N_ * 16);
  const size_t o_r3    = take((size_t)B_ * 3 * N_ * 16);
  const size_t o_idx16 = take((size_t)B_ * N_ * 16);
  const size_t o_d16   = take((size_t)B_ * N_ * 16);
  const size_t o_idx3  = take((size_t)B_ * N_ * 3);
  const size_t o_d3    = take((size_t)B_ * N_ * 3);
  const size_t o_idxd  = take((size_t)B_ * N2_ * 16);
  const size_t o_dd    = take((size_t)B_ * N2_ * 16);

  float* p1    = WS + o_p1;
  float* p2    = WS + o_p2;
  float* x0    = WS + o_x0;
  float* stats = WS + o_stats;
  float* nk    = WS + o_nk;
  float* nv    = WS + o_nv;
  float* nr    = WS + o_nr;
  float* aa    = WS + o_aa;
  float* r3    = WS + o_r3;
  int*   idx16 = (int*)(WS + o_idx16);
  float* d16   = WS + o_d16;
  int*   idx3  = (int*)(WS + o_idx3);
  float* d3    = WS + o_d3;
  int*   idxd  = (int*)(WS + o_idxd);
  float* ddk   = WS + o_dd;
  auto ACT = [&](int i) { return WS + o_act + (size_t)i * B_ * H_ * N_; };

  const float* pc = P(PPC);

  // -- helpers --------------------------------------------------------------
  auto gemm = [&](const float* W, const float* X, const float* bias, float* D,
                  int K, int L, int CX, int M, int MD) {
    const dim3 g(L / 128, 1, B_);
    const int kT = (K + 31) / 32;
    if (M == 64) {
      if (kT == 1)
        k_gemm_wmma<1, true><<<g, dim3(256), 0, stream>>>(W, X, bias, D, K, L, CX, M, MD);
      else if (kT == 2)
        k_gemm_wmma<2, true><<<g, dim3(256), 0, stream>>>(W, X, bias, D, K, L, CX, M, MD);
      else
        k_gemm_wmma<3, true><<<g, dim3(256), 0, stream>>>(W, X, bias, D, K, L, CX, M, MD);
    } else {
      if (kT == 1)
        k_gemm_wmma<1, false><<<g, dim3(256), 0, stream>>>(W, X, bias, D, K, L, CX, M, MD);
      else if (kT == 2)
        k_gemm_wmma<2, false><<<g, dim3(256), 0, stream>>>(W, X, bias, D, K, L, CX, M, MD);
      else
        k_gemm_wmma<3, false><<<g, dim3(256), 0, stream>>>(W, X, bias, D, K, L, CX, M, MD);
    }
  };
  auto bn = [&](const float* X, const float* g, const float* bt,
                const float* res, float* Y, int C, int L, int relu) {
    k_bn_stats<<<dim3(C), dim3(256), 0, stream>>>(X, g, bt, stats, C, L, B_);
    k_bn_apply<<<dim3(nb256(L), C, B_), dim3(256), 0, stream>>>(
        X, stats, res, Y, C, L, relu);
  };

  // -- pt_layer -------------------------------------------------------------
  auto pt_layer = [&](int tr, const float* p, int Npt, const float* xin,
                      float* out) {
    const int L = Npt * 16;
    float* q  = ACT(4);
    float* kk = ACT(5);
    float* vv = ACT(6);
    gemm(P(tr + 1), xin, P(tr + 11), q,  64, Npt, 64, 64, 64);   // Wq,bq
    gemm(P(tr + 0), xin, P(tr + 10), kk, 64, Npt, 64, 64, 64);   // Wk,bk
    gemm(P(tr + 2), xin, P(tr + 12), vv, 64, Npt, 64, 64, 64);   // Wv,bv
    k_knn<16><<<dim3(Npt / 256, 1, B_), dim3(256), 0, stream>>>(
        p, p, idx16, d16, Npt, Npt);
    k_group_xyz<<<dim3(nb256(L), 1, B_), dim3(256), 0, stream>>>(
        p, p, idx16, nk, Npt, Npt, 67);
    k_group_feat<<<dim3(nb256(L), 64, B_), dim3(256), 0, stream>>>(
        kk, idx16, nk, 64, 67, 3, Npt, Npt);
    k_group_feat<<<dim3(nb256(L), 64, B_), dim3(256), 0, stream>>>(
        vv, idx16, nv, 64, 64, 0, Npt, Npt);
    // positional branch: r = relu(bn(pW1 * nk[:, :3])) ; nr = pW2 * r + pb2
    gemm(P(tr + 13), nk, nullptr, r3, 3, L, 67, 3, 3);
    bn(r3, P(tr + 17), P(tr + 16), nullptr, r3, 3, L, 1);
    gemm(P(tr + 14), r3, P(tr + 15), nr, 3, L, 3, 64, 64);
    const long long tfeat = (long long)B_ * 64 * L;
    k_add_inplace<<<dim3(nb256(tfeat)), dim3(256), 0, stream>>>(nv, nr, tfeat);
    // attention logits
    k_attn_form<<<dim3(nb256(L), 64, B_), dim3(256), 0, stream>>>(
        q, nk, nr, aa, 64, 67, Npt);
    bn(aa, P(tr + 7), P(tr + 6), nullptr, aa, 64, L, 1);         // abn1
    gemm(P(tr + 3), aa, nullptr, nr, 64, L, 64, 64, 64);         // aW1 (nr reused)
    bn(nr, P(tr + 9), P(tr + 8), nullptr, nr, 64, L, 1);         // abn2
    gemm(P(tr + 4), nr, P(tr + 5), aa, 64, L, 64, 64, 64);       // aW2,ab2
    const long long tout = (long long)B_ * 64 * Npt;
    k_softmax_dot<<<dim3(nb256(tout)), dim3(256), 0, stream>>>(aa, nv, out, tout);
  };

  // -- residual transformer block ------------------------------------------
  auto block = [&](int bi, const float* p, int Npt, const float* xin,
                   float* xout) {
    float* y  = ACT(3);
    float* pt = ACT(7);
    gemm(P(bi + 0), xin, nullptr, y, 64, Npt, 64, 64, 64);
    bn(y, P(bi + 3), P(bi + 2), nullptr, y, 64, Npt, 1);
    pt_layer(bi + 8, p, Npt, y, pt);
    bn(pt, P(bi + 7), P(bi + 6), nullptr, pt, 64, Npt, 1);
    gemm(P(bi + 1), pt, nullptr, y, 64, Npt, 64, 64, 64);
    bn(y, P(bi + 5), P(bi + 4), xin, xout, 64, Npt, 1);   // relu(bn(.) + x)
  };

  // ======================= forward pass ====================================
  // split input
  k_split_p<<<dim3(nb256((long long)B_ * N_)), dim3(256), 0, stream>>>(
      pc, p1, (long long)B_ * N_);
  k_split_x<<<dim3(nb256(N_), 16, B_), dim3(256), 0, stream>>>(pc, x0, N_);

  // input MLP
  gemm(P(PIN + 0), x0, nullptr, ACT(0), 16, N_, 16, 64, 64);
  bn(ACT(0), P(PIN + 3), P(PIN + 2), nullptr, ACT(0), 64, N_, 1);
  gemm(P(PIN + 1), ACT(0), nullptr, ACT(1), 64, N_, 64, 64, 64);
  bn(ACT(1), P(PIN + 5), P(PIN + 4), nullptr, ACT(1), 64, N_, 1);

  // block b1  (x1 = ACT(2))
  block(PB1, p1, N_, ACT(1), ACT(2));

  // down: fps + knn + group + 2x conv2d + max_k  (x2 = ACT(8))
  k_fps<<<dim3(B_), dim3(256), 0, stream>>>(p1, p2, N_, N2_);
  k_knn<16><<<dim3(N2_ / 256, 1, B_), dim3(256), 0, stream>>>(
      p2, p1, idxd, ddk, N2_, N_);
  {
    const int L2K = N2_ * 16;
    k_group_xyz<<<dim3(nb256(L2K), 1, B_), dim3(256), 0, stream>>>(
        p1, p2, idxd, nk, N2_, N_, 67);
    k_group_feat<<<dim3(nb256(L2K), 64, B_), dim3(256), 0, stream>>>(
        ACT(2), idxd, nk, 64, 67, 3, N2_, N_);
    gemm(P(PDW + 0), nk, nullptr, nv, 67, L2K, 67, 64, 64);
    bn(nv, P(PDW + 3), P(PDW + 2), nullptr, nv, 64, L2K, 1);
    gemm(P(PDW + 1), nv, nullptr, nr, 64, L2K, 64, 64, 64);
    bn(nr, P(PDW + 5), P(PDW + 4), nullptr, nr, 64, L2K, 1);
    const long long tmax = (long long)B_ * 64 * N2_;
    k_max_k<<<dim3(nb256(tmax)), dim3(256), 0, stream>>>(nr, ACT(8), tmax);
  }

  // block b2 at coarse level  (x2b = ACT(9))
  block(PB2, p2, N2_, ACT(8), ACT(9));

  // up: 3-NN interpolation
  k_knn<3><<<dim3(N_ / 256, 1, B_), dim3(256), 0, stream>>>(
      p1, p2, idx3, d3, N_, N2_);
  gemm(P(PUP + 0), ACT(9), nullptr, ACT(10), 64, N2_, 64, 64, 64);   // f1
  bn(ACT(10), P(PUP + 3), P(PUP + 2), nullptr, ACT(10), 64, N2_, 1);
  gemm(P(PUP + 1), ACT(2), nullptr, ACT(11), 64, N_, 64, 64, 64);    // f2
  bn(ACT(11), P(PUP + 5), P(PUP + 4), nullptr, ACT(11), 64, N_, 1);
  k_up_interp<<<dim3(nb256(N_), 64, B_), dim3(256), 0, stream>>>(
      ACT(10), ACT(11), idx3, d3, ACT(12), 64, N_, N2_);

  // block b3  (y3 = ACT(13))
  block(PB3, p1, N_, ACT(12), ACT(13));

  // head
  gemm(P(POUT + 0), ACT(13), nullptr, ACT(14), 64, N_, 64, 64, 64);
  bn(ACT(14), P(POUT + 4), P(POUT + 3), nullptr, ACT(14), 64, N_, 1);
  gemm(P(POUT + 1), ACT(14), P(POUT + 2), (float*)d_out, 64, N_, 64, NCLS_, NCLS_);
}
